// PPOActorCritic_23673859736113
// MI455X (gfx1250) — matrix operand order repaired and checked by reading, not compiled, not run on hardware
//
#include <hip/hip_runtime.h>
#include <hip/hip_bf16.h>
#include <stdint.h>

// ---------------------------------------------------------------------------
// GAT actor-critic for MI455X (gfx1250, wave32).
//   B=16, N=2048, F_IN=6, H=64.
// Pipeline:
//   k_pack_adj : adj int32 [B,N,N] (256MB) -> bitmask (8MB). HBM read once.
//   k_feat*    : h = x@W, stored feature-major f32 (residual) + f16 (WMMA B);
//                s = h@a_src; d = h@a_dst with column mask folded as -inf.
//   k_dmax     : per-batch max of d -> safe softmax row-max bound (leaky_relu
//                is monotonic, so leaky(s_i + Dmax_b) >= every score of row i).
//   k_attn     : streaming masked softmax (single pass, no rescale) +
//                alpha@h via v_wmma_f32_16x16x32_f16, residual + elu + mask.
//   k_actor / k_critic : small MLP heads.
// ---------------------------------------------------------------------------

#define B_   16
#define N_   2048
#define FIN_ 6
#define H_   64

typedef __attribute__((ext_vector_type(16))) _Float16 v16h;
typedef __attribute__((ext_vector_type(8)))  _Float16 v8h;
typedef __attribute__((ext_vector_type(8)))  float    v8f;

#define LOG2E_ 1.44269504088896f

__device__ __forceinline__ float leaky02(float x) { return x > 0.f ? x : 0.2f * x; }
// Raw v_exp_f32. Args here are <= 0 (softmax) or elu (underflow->0 is correct).
__device__ __forceinline__ float fastexp(float x) {
  return __builtin_amdgcn_exp2f(x * LOG2E_);
}
__device__ __forceinline__ float elu1(float x) { return x > 0.f ? x : fastexp(x) - 1.f; }

// ---------------------------------------------------------------------------
// Pack adjacency to bits: one thread per 32 entries (128B contiguous per lane).
__global__ __launch_bounds__(256) void k_pack_adj(const int* __restrict__ adj,
                                                  uint32_t* __restrict__ bits) {
  size_t w = (size_t)blockIdx.x * blockDim.x + threadIdx.x;   // word index
  const int4* p = (const int4*)adj + w * 8;
  uint32_t m = 0;
#pragma unroll
  for (int q = 0; q < 8; ++q) {
    int4 v = p[q];
    m |= (v.x != 0 ? 1u : 0u) << (q * 4 + 0);
    m |= (v.y != 0 ? 1u : 0u) << (q * 4 + 1);
    m |= (v.z != 0 ? 1u : 0u) << (q * 4 + 2);
    m |= (v.w != 0 ? 1u : 0u) << (q * 4 + 3);
  }
  bits[w] = m;
}

// ---------------------------------------------------------------------------
// Feature transform, layer 1 (F_IN=6). One thread per node.
__global__ __launch_bounds__(256) void k_feat1(
    const float* __restrict__ x, const float* __restrict__ W,
    const float* __restrict__ asrc, const float* __restrict__ adst,
    const float* __restrict__ mask,
    float* __restrict__ hT32, _Float16* __restrict__ hT16,
    float* __restrict__ s, float* __restrict__ d) {
  __shared__ float Wl[FIN_ * H_];
  __shared__ float As[H_], Ad[H_];
  int t = threadIdx.x;
  for (int q = t; q < FIN_ * H_; q += blockDim.x) Wl[q] = W[q];
  if (t < H_) { As[t] = asrc[t]; Ad[t] = adst[t]; }
  __syncthreads();
  int node = blockIdx.x * blockDim.x + t;       // [0, B*N)
  int b = node >> 11, col = node & (N_ - 1);
  float xv[FIN_];
#pragma unroll
  for (int c = 0; c < FIN_; ++c) xv[c] = x[(size_t)node * FIN_ + c];
  float sv = 0.f, dv = 0.f;
#pragma unroll
  for (int f = 0; f < H_; ++f) {
    float acc = 0.f;
#pragma unroll
    for (int c = 0; c < FIN_; ++c) acc = fmaf(xv[c], Wl[c * H_ + f], acc);
    size_t o = ((size_t)b * H_ + f) * N_ + col; // feature-major -> coalesced
    hT32[o] = acc;
    hT16[o] = (_Float16)acc;
    sv = fmaf(acc, As[f], sv);
    dv = fmaf(acc, Ad[f], dv);
  }
  s[node] = sv;
  d[node] = (mask[node] > 0.f) ? dv : -__builtin_inff();   // fold column mask
}

// Feature transform, layer 2 (F_IN=64). One thread per node, W2 in LDS.
__global__ __launch_bounds__(256) void k_feat2(
    const float* __restrict__ gin, const float* __restrict__ W,
    const float* __restrict__ asrc, const float* __restrict__ adst,
    const float* __restrict__ mask,
    float* __restrict__ hT32, _Float16* __restrict__ hT16,
    float* __restrict__ s, float* __restrict__ d) {
  __shared__ float Wl[H_ * H_];
  __shared__ float As[H_], Ad[H_];
  int t = threadIdx.x;
  for (int q = t; q < H_ * H_; q += blockDim.x) Wl[q] = W[q];
  if (t < H_) { As[t] = asrc[t]; Ad[t] = adst[t]; }
  __syncthreads();
  int node = blockIdx.x * blockDim.x + t;
  int b = node >> 11, col = node & (N_ - 1);
  float xv[H_];
  const float4* gp = (const float4*)(gin + (size_t)node * H_);
#pragma unroll
  for (int q = 0; q < H_ / 4; ++q) {
    float4 v = gp[q];
    xv[4 * q + 0] = v.x; xv[4 * q + 1] = v.y; xv[4 * q + 2] = v.z; xv[4 * q + 3] = v.w;
  }
  float sv = 0.f, dv = 0.f;
  for (int f = 0; f < H_; ++f) {
    float acc = 0.f;
#pragma unroll
    for (int c = 0; c < H_; ++c) acc = fmaf(xv[c], Wl[c * H_ + f], acc);  // LDS broadcast
    size_t o = ((size_t)b * H_ + f) * N_ + col;
    hT32[o] = acc;
    hT16[o] = (_Float16)acc;
    sv = fmaf(acc, As[f], sv);
    dv = fmaf(acc, Ad[f], dv);
  }
  s[node] = sv;
  d[node] = (mask[node] > 0.f) ? dv : -__builtin_inff();
}

// ---------------------------------------------------------------------------
// Per-batch max of d (column mask already folded in as -inf).
__global__ __launch_bounds__(256) void k_dmax(const float* __restrict__ d,
                                              float* __restrict__ Dmax) {
  __shared__ float red[256];
  int b = blockIdx.x, t = threadIdx.x;
  float m = -__builtin_inff();
  for (int n = t; n < N_; n += 256) m = fmaxf(m, d[b * N_ + n]);
  red[t] = m;
  __syncthreads();
  for (int w = 128; w > 0; w >>= 1) {
    if (t < w) red[t] = fmaxf(red[t], red[t + w]);
    __syncthreads();
  }
  if (t == 0) Dmax[b] = red[0];
}

// ---------------------------------------------------------------------------
// Streaming masked-softmax attention + alpha@h (WMMA) + residual + elu + mask.
// Block = (batch b, 16-row tile). 128 threads = 4 waves; wave w owns output
// columns [16w, 16w+16). Thread layout for the p-tile: lane = column k (fixed),
// wave w computes rows {w, w+4, w+8, w+12} -> s/Mi/bit-row pointers are
// loop-invariant registers, d is one coalesced load per tile, p is branchless.
// Softmax denominators accumulate per-thread and reduce once via shfl_xor.
// 2-way jt unroll: ping-pong p-tile buffers (1 barrier/tile) + two independent
// WMMA accumulators (breaks the D->C RAW chain between consecutive WMMAs).
__global__ __launch_bounds__(128) void k_attn(
    const float* __restrict__ s, const float* __restrict__ d,
    const float* __restrict__ Dmax, const uint32_t* __restrict__ bits,
    const _Float16* __restrict__ hT16, const float* __restrict__ hT32,
    const float* __restrict__ mask, float* __restrict__ g) {
  __shared__ _Float16 pt[2][16 * 32];  // ping-pong p tiles, row-major [i][k]
  __shared__ float lrow[16], mrow[16];
  __shared__ float osh[16 * H_];

  const int t    = threadIdx.x;
  const int lane = t & 31;             // p-tile column k
  const int wave = t >> 5;
  const int b    = blockIdx.x >> 7;    // 128 row-tiles per batch
  const int i0   = (blockIdx.x & 127) << 4;

  // Loop-invariant per-thread row state (rows wave + 4q).
  const float DM = Dmax[b];
  float srw[4], MiS[4];                // MiS = leaky(s+Dmax) * log2(e)
  const uint32_t* brow[4];
#pragma unroll
  for (int q = 0; q < 4; ++q) {
    int row = wave + 4 * q;
    float sv = s[b * N_ + i0 + row];
    srw[q] = sv;
    MiS[q] = leaky02(sv + DM) * LOG2E_;
    brow[q] = bits + ((size_t)b * N_ + i0 + row) * (N_ / 32);
  }
  if (t < 16) mrow[t] = mask[b * N_ + i0 + t];

  const float* dptr = d + b * N_ + lane;

  float lsum[4] = {0.f, 0.f, 0.f, 0.f};
  v8f acc0 = {}, acc1 = {};

  // WMMA fragment geometry (per CDNA5 ISA layouts).
  const int c0  = wave << 4;
  const int M   = lane & 15;
  const int k0a = (lane >> 4) << 3;    // A frag: K chunks of 8 per lane-half
  const int k0b = (lane >> 4) << 3;    // B frag: K chunks of 8 per lane-half
  const _Float16* hbase = hT16 + ((size_t)b * H_ + c0 + M) * N_;

#pragma unroll 1
  for (int jt = 0; jt < N_ / 32; jt += 2) {
#pragma unroll
    for (int u = 0; u < 2; ++u) {
      const int jti = jt + u;
      const int j   = jti << 5;
      _Float16* ptb = pt[u];
      v8f& acc = u ? acc1 : acc0;

      // ---- p tile: p(i,k) = adjbit * exp2(leaky(s_i + d_k)*log2e - MiS_i)
      float dv = dptr[j];              // coalesced b32; -inf if column masked
#pragma unroll
      for (int q = 0; q < 4; ++q) {
        uint32_t w32 = brow[q][jti];
        __builtin_prefetch(&brow[q][jti + 2], 0, 3);
        float sel = (float)((w32 >> lane) & 1u);
        float e   = leaky02(srw[q] + dv);
        float p   = sel * __builtin_amdgcn_exp2f(fmaf(e, LOG2E_, -MiS[q]));
        lsum[q] += p;
        ptb[(wave + 4 * q) * 32 + lane] = (_Float16)p;
      }
      __syncthreads();                 // p tile visible; prev buf reads done

      // ---- A fragment: row M, halves K=[k0a..k0a+7] and [k0a+16..k0a+23]
      v8h alo = *(const v8h*)&ptb[M * 32 + k0a];
      v8h ahi = *(const v8h*)&ptb[M * 32 + k0a + 16];
      v16h A;
#pragma unroll
      for (int q = 0; q < 8; ++q) { A[q] = alo[q]; A[q + 8] = ahi[q]; }

      // ---- B fragment: column c0+M, halves K=[k0b..k0b+7] and [k0b+16..k0b+23]
      const _Float16* bp = hbase + j + k0b;
      v8h blo = *(const v8h*)bp;
      v8h bhi = *(const v8h*)(bp + 16);
      v16h Bv;
#pragma unroll
      for (int q = 0; q < 8; ++q) { Bv[q] = blo[q]; Bv[q + 8] = bhi[q]; }

      acc = __builtin_amdgcn_wmma_f32_16x16x32_f16(false, A, false, Bv,
                                                   (short)0, acc, false, false);
    }
  }

  // ---- softmax denominators: wave-reduce each of the 4 row partials
#pragma unroll
  for (int q = 0; q < 4; ++q) {
    float v = lsum[q];
#pragma unroll
    for (int off = 16; off > 0; off >>= 1) v += __shfl_xor(v, off, 32);
    if (lane == 0) lrow[wave + 4 * q] = v;
  }
  __syncthreads();

  // ---- epilogue: out = elu(O/l + h) * mask_row
  v8f acc = acc0 + acc1;
#pragma unroll
  for (int r = 0; r < 8; ++r) {
    int Mr = r + ((lane >> 4) << 3);   // D layout: VGPR r -> row r+8*(lane>=16)
    int f  = c0 + (lane & 15);
    float li = lrow[Mr];
    float hi = hT32[((size_t)b * H_ + f) * N_ + i0 + Mr];
    float v  = (li > 0.f) ? (acc[r] / li + hi) : hi;  // l==0 -> nan_to_num path
    osh[Mr * H_ + f] = elu1(v) * mrow[Mr];
  }
  __syncthreads();
#pragma unroll
  for (int q = 0; q < 8; ++q) {        // coalesced row-major store
    int e = t + q * 128;
    g[((size_t)b * N_ + i0) * H_ + e] = osh[e];
  }
}

// ---------------------------------------------------------------------------
// Actor head: logits = elu(g@W1 + b1) @ W2 + b2, one thread per node.
__global__ __launch_bounds__(256) void k_actor(
    const float* __restrict__ g, const float* __restrict__ w1,
    const float* __restrict__ b1, const float* __restrict__ w2,
    const float* __restrict__ b2, float* __restrict__ logits) {
  __shared__ float W1l[H_ * 32];
  __shared__ float B1l[32], W2l[32];
  int t = threadIdx.x;
  for (int q = t; q < H_ * 32; q += blockDim.x) W1l[q] = w1[q];
  if (t < 32) { B1l[t] = b1[t]; W2l[t] = w2[t]; }
  __syncthreads();
  int node = blockIdx.x * blockDim.x + t;
  float xv[H_];
  const float4* gp = (const float4*)(g + (size_t)node * H_);
#pragma unroll
  for (int q = 0; q < H_ / 4; ++q) {
    float4 v = gp[q];
    xv[4 * q + 0] = v.x; xv[4 * q + 1] = v.y; xv[4 * q + 2] = v.z; xv[4 * q + 3] = v.w;
  }
  float out = b2[0];
  for (int o = 0; o < 32; ++o) {
    float a = B1l[o];
#pragma unroll
    for (int c = 0; c < H_; ++c) a = fmaf(xv[c], W1l[c * 32 + o], a);
    out = fmaf(elu1(a), W2l[o], out);
  }
  logits[node] = out;
}

// Critic head: per-batch mean-pool (masked count) + 64->32->1 MLP.
__global__ __launch_bounds__(256) void k_critic(
    const float* __restrict__ g, const float* __restrict__ mask,
    const float* __restrict__ w1, const float* __restrict__ b1,
    const float* __restrict__ w2, const float* __restrict__ b2,
    float* __restrict__ value) {
  __shared__ float red[256];
  __shared__ float sumh[H_];
  __shared__ float c1[32];
  __shared__ float cnt;
  int b = blockIdx.x, t = threadIdx.x;
  int f = t & 63, c = t >> 6;                     // 64 features x 4 chunks
  float a = 0.f;
  for (int n = c; n < N_; n += 4) a += g[((size_t)b * N_ + n) * H_ + f];
  float mc = 0.f;
  for (int n = t; n < N_; n += 256) mc += mask[b * N_ + n];
  red[t] = a;
  __syncthreads();
  if (t < 64) sumh[t] = red[t] + red[t + 64] + red[t + 128] + red[t + 192];
  __syncthreads();
  red[t] = mc;
  __syncthreads();
  for (int w = 128; w > 0; w >>= 1) {
    if (t < w) red[t] += red[t + w];
    __syncthreads();
  }
  if (t == 0) cnt = fmaxf(red[0], 1.f);
  __syncthreads();
  if (t < 32) {
    float a2 = b1[t];
#pragma unroll
    for (int q = 0; q < H_; ++q) a2 = fmaf(sumh[q] / cnt, w1[q * 32 + t], a2);
    c1[t] = elu1(a2) * w2[t];
  }
  __syncthreads();
  if (t == 0) {
    float v = b2[0];
#pragma unroll
    for (int q = 0; q < 32; ++q) v += c1[q];
    value[b] = v;
  }
}

// ---------------------------------------------------------------------------
extern "C" void kernel_launch(void* const* d_in, const int* in_sizes, int n_in,
                              void* d_out, int out_size, void* d_ws, size_t ws_size,
                              hipStream_t stream) {
  (void)in_sizes; (void)n_in; (void)out_size; (void)ws_size;
  const float* x    = (const float*)d_in[0];
  const int*   adj  = (const int*)d_in[1];
  const float* mask = (const float*)d_in[2];
  const float* W1   = (const float*)d_in[3];
  const float* as1  = (const float*)d_in[4];
  const float* ad1  = (const float*)d_in[5];
  const float* W2   = (const float*)d_in[6];
  const float* as2  = (const float*)d_in[7];
  const float* ad2  = (const float*)d_in[8];
  const float* aw1  = (const float*)d_in[9];
  const float* ab1  = (const float*)d_in[10];
  const float* aw2  = (const float*)d_in[11];
  const float* ab2  = (const float*)d_in[12];
  const float* cw1  = (const float*)d_in[13];
  const float* cb1  = (const float*)d_in[14];
  const float* cw2  = (const float*)d_in[15];
  const float* cb2  = (const float*)d_in[16];
  float* logits = (float*)d_out;
  float* value  = (float*)d_out + (size_t)B_ * N_;

  char* ws = (char*)d_ws;
  size_t off = 0;
  auto alloc = [&](size_t bytes) -> void* {
    void* p = ws + off;
    off = (off + bytes + 255) & ~(size_t)255;
    return p;
  };
  uint32_t* bits = (uint32_t*)alloc((size_t)B_ * N_ * (N_ / 32) * 4); // 8 MB
  _Float16* hT16 = (_Float16*)alloc((size_t)B_ * H_ * N_ * 2);        // 4 MB
  float*    hT32 = (float*)   alloc((size_t)B_ * H_ * N_ * 4);        // 8 MB
  float*    sbuf = (float*)   alloc((size_t)B_ * N_ * 4);
  float*    dbuf = (float*)   alloc((size_t)B_ * N_ * 4);
  float*    Dmx  = (float*)   alloc((size_t)B_ * 4);
  float*    g    = (float*)   alloc((size_t)B_ * N_ * H_ * 4);        // 8 MB (reused)

  const int words = B_ * N_ * (N_ / 32);
  k_pack_adj<<<words / 256, 256, 0, stream>>>(adj, bits);

  // Layer 1
  k_feat1<<<(B_ * N_) / 256, 256, 0, stream>>>(x, W1, as1, ad1, mask, hT32, hT16, sbuf, dbuf);
  k_dmax <<<B_, 256, 0, stream>>>(dbuf, Dmx);
  k_attn <<<B_ * (N_ / 16), 128, 0, stream>>>(sbuf, dbuf, Dmx, bits, hT16, hT32, mask, g);

  // Layer 2 (h-buffers reused; attn2 writes g after feat2 consumed it)
  k_feat2<<<(B_ * N_) / 256, 256, 0, stream>>>(g, W2, as2, ad2, mask, hT32, hT16, sbuf, dbuf);
  k_dmax <<<B_, 256, 0, stream>>>(dbuf, Dmx);
  k_attn <<<B_ * (N_ / 16), 128, 0, stream>>>(sbuf, dbuf, Dmx, bits, hT16, hT32, mask, g);

  // Heads
  k_actor <<<(B_ * N_) / 256, 256, 0, stream>>>(g, aw1, ab1, aw2, ab2, logits);
  k_critic<<<B_, 256, 0, stream>>>(g, mask, cw1, cb1, cw2, cb2, value);
}
